// AutoCorrelation_1726576856534
// MI455X (gfx1250) — matrix operand order repaired and checked by reading, not compiled, not run on hardware
//
#include <hip/hip_runtime.h>

#define B_SZ   8
#define L_SEQ  1536
#define CH     1024      // H*E = 16*64
#define NST    48        // L_SEQ / 32 super-tiles
#define TOPK   7         // int(log(1536))

typedef __attribute__((ext_vector_type(16))) __bf16 v16bf;
typedef __attribute__((ext_vector_type(8)))  float  v8f;

static __device__ __forceinline__ unsigned short f32_to_bf16_rne(float f) {
    unsigned int u = __float_as_uint(f);
    u += 0x7FFFu + ((u >> 16) & 1u);   // round-to-nearest-even
    return (unsigned short)(u >> 16);
}

// ---------------- Phase 0: convert Q,K f32 -> bf16 (into d_out scratch), zero mean bins
__global__ __launch_bounds__(256) void ac_convert(
    const float* __restrict__ q, const float* __restrict__ k,
    unsigned short* __restrict__ qb, unsigned short* __restrict__ kb,
    float* __restrict__ meanv, int n4)
{
    int i = blockIdx.x * 256 + threadIdx.x;
    if (i < B_SZ * L_SEQ) meanv[i] = 0.0f;
    if (i >= n4) return;
    float4 qv = ((const float4*)q)[i];
    float4 kv = ((const float4*)k)[i];
    ushort4 qo, ko;
    qo.x = f32_to_bf16_rne(qv.x); qo.y = f32_to_bf16_rne(qv.y);
    qo.z = f32_to_bf16_rne(qv.z); qo.w = f32_to_bf16_rne(qv.w);
    ko.x = f32_to_bf16_rne(kv.x); ko.y = f32_to_bf16_rne(kv.y);
    ko.z = f32_to_bf16_rne(kv.z); ko.w = f32_to_bf16_rne(kv.w);
    ((ushort4*)qb)[i] = qo;
    ((ushort4*)kb)[i] = ko;
}

// ---------------- Phase 1: mean_value[b,tau] = (1/1024) * sum_t <Q[t+tau], K[t]>
// One workgroup = (batch b, super-diagonal band Delta): I - J == 32*Delta (mod L).
// 8 waves sweep the 48 super-tiles of the band. Each wave computes a 32x32
// G super-tile as a 2x2 register block of V_WMMA_F32_16X16X32_BF16 tiles:
// 2 A-fragments + 2 B-fragments feed 4 WMMAs per k-step (halves load traffic
// per WMMA vs unblocked, 4 independent accumulation chains). Results fold into
// 63 circular-diagonal LDS bins, then 63 global atomics per workgroup.
__global__ __launch_bounds__(256) void ac_corr_wmma(
    const unsigned short* __restrict__ qb,
    const unsigned short* __restrict__ kb,
    float* __restrict__ meanv)
{
    __shared__ float diag[63];
    const int b     = blockIdx.x / NST;
    const int Delta = blockIdx.x % NST;
    const int tid   = threadIdx.x;
    const int wave  = tid >> 5;
    const int lane  = tid & 31;
    const int m     = lane & 15;
    const int half  = lane >> 4;

    if (tid < 63) diag[tid] = 0.0f;
    __syncthreads();

    const unsigned short* qbase = qb + (size_t)b * L_SEQ * CH;
    const unsigned short* kbase = kb + (size_t)b * L_SEQ * CH;

    for (int sj = wave; sj < NST; sj += 8) {
        int si = sj + Delta; if (si >= NST) si -= NST;
        const int I = si << 5, J = sj << 5;
        const unsigned short* qr0 = qbase + (size_t)(I + m)      * CH;
        const unsigned short* qr1 = qbase + (size_t)(I + 16 + m) * CH;
        const unsigned short* kr0 = kbase + (size_t)(J + m)      * CH;
        const unsigned short* kr1 = kbase + (size_t)(J + 16 + m) * CH;

        v8f a00 = {0.f,0.f,0.f,0.f,0.f,0.f,0.f,0.f};
        v8f a01 = a00, a10 = a00, a11 = a00;

        #pragma unroll 2
        for (int kc = 0; kc < CH; kc += 32) {
            union { uint4 u[2]; v16bf v; } A0, A1, B0, B1;
            // A packing: element e -> K = (e>>3)*16 + half*8 + (e&7)
            A0.u[0] = *(const uint4*)(qr0 + kc + half * 8);
            A0.u[1] = *(const uint4*)(qr0 + kc + 16 + half * 8);
            A1.u[0] = *(const uint4*)(qr1 + kc + half * 8);
            A1.u[1] = *(const uint4*)(qr1 + kc + 16 + half * 8);
            // B packing: element e -> K = (e>>3)*16 + half*8 + (e&7)
            B0.u[0] = *(const uint4*)(kr0 + kc + half * 8);
            B0.u[1] = *(const uint4*)(kr0 + kc + 16 + half * 8);
            B1.u[0] = *(const uint4*)(kr1 + kc + half * 8);
            B1.u[1] = *(const uint4*)(kr1 + kc + 16 + half * 8);
            a00 = __builtin_amdgcn_wmma_f32_16x16x32_bf16(
                      false, A0.v, false, B0.v, (short)0, a00, false, false);
            a01 = __builtin_amdgcn_wmma_f32_16x16x32_bf16(
                      false, A0.v, false, B1.v, (short)0, a01, false, false);
            a10 = __builtin_amdgcn_wmma_f32_16x16x32_bf16(
                      false, A1.v, false, B0.v, (short)0, a10, false, false);
            a11 = __builtin_amdgcn_wmma_f32_16x16x32_bf16(
                      false, A1.v, false, B1.v, (short)0, a11, false, false);
        }

        // D layout: acc[r] = G[row_base + r + 8*half, col_base + n], n = lane&15.
        // Within the super-tile, diag offset o = (ti - tj) + r + 8*half - m,
        // ti,tj in {0,16}; tiles (0,0) and (16,16) share a bin.
        #pragma unroll
        for (int r = 0; r < 8; ++r) {
            int o = r + (half << 3) - m;               // [-15, 15]
            atomicAdd(&diag[o + 31],      a00[r] + a11[r]);
            atomicAdd(&diag[o + 31 - 16], a01[r]);     // ti=0,  tj=16
            atomicAdd(&diag[o + 31 + 16], a10[r]);     // ti=16, tj=0
        }
    }
    __syncthreads();

    if (tid < 63) {
        int tau = Delta * 32 + tid - 31;
        if (tau < 0) tau += L_SEQ; else if (tau >= L_SEQ) tau -= L_SEQ;
        atomicAdd(&meanv[b * L_SEQ + tau], diag[tid] * (1.0f / 1024.0f));
    }
}

// ---------------- Phase 2: batch-mean top-7 lags + per-batch softmax weights
__global__ __launch_bounds__(256) void ac_topk_softmax(
    const float* __restrict__ meanv, int* __restrict__ idx_out,
    float* __restrict__ w_out)
{
    __shared__ float sv[L_SEQ];
    __shared__ float rv[256];
    __shared__ int   ri[256];
    __shared__ int   sidx[TOPK];
    const int tid = threadIdx.x;

    for (int t = tid; t < L_SEQ; t += 256) {
        float s = 0.0f;
        #pragma unroll
        for (int b = 0; b < B_SZ; ++b) s += meanv[b * L_SEQ + t];
        sv[t] = s;
    }
    __syncthreads();

    for (int k = 0; k < TOPK; ++k) {
        float best = -INFINITY; int bestI = 0;
        for (int t = tid; t < L_SEQ; t += 256)
            if (sv[t] > best) { best = sv[t]; bestI = t; }
        rv[tid] = best; ri[tid] = bestI;
        __syncthreads();
        for (int s = 128; s > 0; s >>= 1) {
            if (tid < s && rv[tid + s] > rv[tid]) { rv[tid] = rv[tid + s]; ri[tid] = ri[tid + s]; }
            __syncthreads();
        }
        if (tid == 0) { sidx[k] = ri[0]; sv[ri[0]] = -INFINITY; }
        __syncthreads();
    }

    if (tid < TOPK) idx_out[tid] = sidx[tid];
    if (tid < B_SZ) {
        float w[TOPK], mx = -INFINITY;
        #pragma unroll
        for (int k = 0; k < TOPK; ++k) { w[k] = meanv[tid * L_SEQ + sidx[k]]; mx = fmaxf(mx, w[k]); }
        float sum = 0.0f;
        #pragma unroll
        for (int k = 0; k < TOPK; ++k) { w[k] = expf(w[k] - mx); sum += w[k]; }
        float inv = 1.0f / sum;
        #pragma unroll
        for (int k = 0; k < TOPK; ++k) w_out[tid * TOPK + k] = w[k] * inv;
    }
}

// ---------------- Phase 3: out[b,l,:] = sum_i w[b,i] * values[b,(l+idx_i)%L,:]
__global__ __launch_bounds__(256) void ac_aggregate(
    const float* __restrict__ vals, const int* __restrict__ idx,
    const float* __restrict__ w, float* __restrict__ out)
{
    size_t gid = (size_t)blockIdx.x * 256 + threadIdx.x;   // over B*L*(CH/4)
    const int C4 = CH / 4;
    int c4 = (int)(gid % C4);
    size_t bl = gid / C4;
    int l = (int)(bl % L_SEQ);
    int b = (int)(bl / L_SEQ);
    if (b >= B_SZ) return;

    float4 acc = make_float4(0.f, 0.f, 0.f, 0.f);
    #pragma unroll
    for (int k = 0; k < TOPK; ++k) {
        int lsrc = l + idx[k]; if (lsrc >= L_SEQ) lsrc -= L_SEQ;
        const float4 vv = *(const float4*)(vals + ((size_t)b * L_SEQ + lsrc) * CH + c4 * 4);
        float wk = w[b * TOPK + k];
        acc.x += wk * vv.x; acc.y += wk * vv.y;
        acc.z += wk * vv.z; acc.w += wk * vv.w;
    }
    *(float4*)(out + gid * 4) = acc;
}

extern "C" void kernel_launch(void* const* d_in, const int* in_sizes, int n_in,
                              void* d_out, int out_size, void* d_ws, size_t ws_size,
                              hipStream_t stream) {
    const float* q = (const float*)d_in[0];
    const float* k = (const float*)d_in[1];
    const float* v = (const float*)d_in[2];
    (void)in_sizes; (void)n_in; (void)out_size; (void)ws_size;

    // d_out doubles as bf16 scratch for Q,K (exactly out_size*4 bytes), fully
    // overwritten by ac_aggregate at the end.
    unsigned short* qb = (unsigned short*)d_out;
    unsigned short* kb = qb + (size_t)B_SZ * L_SEQ * CH;

    float* meanv = (float*)d_ws;                                        // 8*1536 f32
    int*   idx   = (int*)((char*)d_ws + B_SZ * L_SEQ * sizeof(float));  // 7 ints (pad 16)
    float* w     = (float*)(idx + 16);                                  // 8*7 f32

    const int n  = B_SZ * L_SEQ * CH;
    const int n4 = n / 4;

    ac_convert<<<(n4 + 255) / 256, 256, 0, stream>>>(q, k, qb, kb, meanv, n4);
    ac_corr_wmma<<<B_SZ * NST, 256, 0, stream>>>(qb, kb, meanv);
    ac_topk_softmax<<<1, 256, 0, stream>>>(meanv, idx, w);
    ac_aggregate<<<(B_SZ * L_SEQ * (CH / 4) + 255) / 256, 256, 0, stream>>>(v, idx, w, (float*)d_out);
}
